// GanaChebConv_27522150433358
// MI455X (gfx1250) — matrix operand order repaired and checked by reading, not compiled, not run on hardware
//
#include <hip/hip_runtime.h>
#include <hip/hip_bf16.h>
#include <math.h>

typedef __attribute__((ext_vector_type(16))) _Float16 v16h;
typedef __attribute__((ext_vector_type(8)))  _Float16 v8h;
typedef __attribute__((ext_vector_type(8)))  float    v8f;

// ---------------------------------------------------------------- utilities

__global__ void zero_f32(float* __restrict__ p, long n) {
    long i = (long)blockIdx.x * blockDim.x + threadIdx.x;
    if (i < n) p[i] = 0.0f;
}

__global__ void deg_count(const int* __restrict__ src, float* __restrict__ deg, int E) {
    int e = blockIdx.x * blockDim.x + threadIdx.x;
    if (e < E) atomicAdd(&deg[src[e]], 1.0f);
}

__global__ void dinv_kernel(const float* __restrict__ deg, float* __restrict__ dinv, int N) {
    int i = blockIdx.x * blockDim.x + threadIdx.x;
    if (i < N) {
        float d = deg[i];
        dinv[i] = (d > 0.0f) ? rsqrtf(fmaxf(d, 1.0f)) : 0.0f;
    }
}

__global__ void wnorm_kernel(const int* __restrict__ src, const int* __restrict__ dst,
                             const float* __restrict__ dinv, float* __restrict__ w, int E) {
    int e = blockIdx.x * blockDim.x + threadIdx.x;
    if (e < E) w[e] = -dinv[src[e]] * dinv[dst[e]];
}

// Convert W[k][128][Oin] (f32 row-major) -> Wt[k][Opad][128] (f16 column-major, zero pad)
__global__ void convert_w(const float* __restrict__ W, _Float16* __restrict__ Wt,
                          int Oin, int Opad, int total) {
    int idx = blockIdx.x * blockDim.x + threadIdx.x;
    if (idx >= total) return;
    int kd = idx & 127;
    int n  = (idx >> 7) % Opad;
    int k  = idx / (Opad << 7);
    float v = (n < Oin) ? W[((size_t)(k * 128 + kd)) * Oin + n] : 0.0f;
    Wt[idx] = (_Float16)v;
}

// ---------------------------------------------------------------- SpMM: Y[dst] += w * X[src], F=128
// 32 threads per edge, float4 per thread. X is 25.6MB -> L2-resident on 192MB L2.
__global__ void spmm_edge(const int* __restrict__ src, const int* __restrict__ dst,
                          const float* __restrict__ w, const float* __restrict__ X,
                          float* __restrict__ Y, int E) {
    long t = (long)blockIdx.x * blockDim.x + threadIdx.x;
    int e = (int)(t >> 5);
    if (e >= E) return;
    int f = ((int)t & 31) * 4;
    float wv = w[e];
    const float4 v = *(const float4*)(X + (size_t)src[e] * 128 + f);
    float* yd = Y + (size_t)dst[e] * 128 + f;
    atomicAdd(yd + 0, wv * v.x);
    atomicAdd(yd + 1, wv * v.y);
    atomicAdd(yd + 2, wv * v.z);
    atomicAdd(yd + 3, wv * v.w);
}

// t = 2*t - h   (Chebyshev recurrence finish)
__global__ void fma2sub(float* __restrict__ t, const float* __restrict__ h, long n) {
    long i = (long)blockIdx.x * blockDim.x + threadIdx.x;
    if (i < n) t[i] = 2.0f * t[i] - h[i];
}

// h = relu(acc)   (bias already folded into the k=0 GEMM accumulator init)
__global__ void relu_k(const float* __restrict__ acc, float* __restrict__ h, long n) {
    long i = (long)blockIdx.x * blockDim.x + threadIdx.x;
    if (i < n) h[i] = fmaxf(acc[i], 0.0f);
}

// ---------------------------------------------------------------- WMMA GEMM
// BETA0=true : C = bias[n] + A*W   (accumulator seeded with bias; no C read, no zero pass)
// BETA0=false: C += A*W
// Each wave owns a 16-row strip: A fragments loaded+converted ONCE (4 x v16h = 32 VGPRs),
// reused across all O/16 column tiles (fully unrolled; 4 WMMAs per tile over K=128).
// Wave-uniform `full` flag keeps the common path free of per-lane EXEC branching;
// EXEC is all-ones at every WMMA (ISA 7.12 requirement).
template <int O, bool BETA0>
__global__ void __launch_bounds__(256)
cheb_gemm_wmma(const float* __restrict__ A, const _Float16* __restrict__ Wt,
               float* __restrict__ C, const float* __restrict__ bias, int biasN, int M) {
    const int wave = threadIdx.x >> 5;
    const int tm   = blockIdx.x * 8 + wave;
    if (tm * 16 >= M) return;                 // wave-uniform exit
    const int lane = threadIdx.x & 31;
    const int l15  = lane & 15;
    const bool hi  = lane >= 16;
    const bool full = (tm * 16 + 16 <= M);    // wave-uniform

    int rowA = tm * 16 + l15;
    if (rowA >= M) rowA = M - 1;
    const float* __restrict__ arow = A + (size_t)rowA * 128;

    // ---- load + convert the 16x128 A strip once ----
    // A frag per 32-K chunk: lanes 0-15 -> K {c+0..7, c+16..23}; lanes 16-31 -> K {c+8..15, c+24..31}
    v16h afr[4];
#pragma unroll
    for (int c4 = 0; c4 < 4; ++c4) {
        const int abase = c4 * 32 + (hi ? 8 : 0);
        const float4 a0 = *(const float4*)(arow + abase);
        const float4 a1 = *(const float4*)(arow + abase + 4);
        const float4 a2 = *(const float4*)(arow + abase + 16);
        const float4 a3 = *(const float4*)(arow + abase + 20);
        v16h a;
        a[0]  = (_Float16)a0.x; a[1]  = (_Float16)a0.y; a[2]  = (_Float16)a0.z; a[3]  = (_Float16)a0.w;
        a[4]  = (_Float16)a1.x; a[5]  = (_Float16)a1.y; a[6]  = (_Float16)a1.z; a[7]  = (_Float16)a1.w;
        a[8]  = (_Float16)a2.x; a[9]  = (_Float16)a2.y; a[10] = (_Float16)a2.z; a[11] = (_Float16)a2.w;
        a[12] = (_Float16)a3.x; a[13] = (_Float16)a3.y; a[14] = (_Float16)a3.z; a[15] = (_Float16)a3.w;
        afr[c4] = a;
    }

    const int cm0 = tm * 16 + (hi ? 8 : 0);

    // ---- sweep all column tiles, reusing the A strip ----
#pragma unroll
    for (int tn = 0; tn < O / 16; ++tn) {
        const _Float16* __restrict__ bcol = Wt + (size_t)(tn * 16 + l15) * 128;
        const int cn = tn * 16 + l15;
        float* __restrict__ crow = C + (size_t)cm0 * O + cn;

        v8f c;
        if (BETA0) {
            const float bv = (cn < biasN) ? bias[cn] : 0.0f;
#pragma unroll
            for (int r = 0; r < 8; ++r) c[r] = bv;
        } else if (full) {
#pragma unroll
            for (int r = 0; r < 8; ++r) c[r] = crow[(size_t)r * O];
        } else {
#pragma unroll
            for (int r = 0; r < 8; ++r) c[r] = (cm0 + r < M) ? crow[(size_t)r * O] : 0.0f;
        }

#pragma unroll
        for (int c4 = 0; c4 < 4; ++c4) {
            // B frag: lane-column n; lanes 0-15 -> K {c+0..7, c+16..23}; lanes 16-31 -> K {c+8..15, c+24..31}
            const int bbase = c4 * 32 + (hi ? 8 : 0);
            const v8h b0 = *(const v8h*)(bcol + bbase);
            const v8h b1 = *(const v8h*)(bcol + bbase + 16);
            v16h b;
#pragma unroll
            for (int i = 0; i < 8; ++i) { b[i] = b0[i]; b[i + 8] = b1[i]; }
            c = __builtin_amdgcn_wmma_f32_16x16x32_f16(false, afr[c4], false, b,
                                                       (short)0, c, false, false);
        }

        if (full) {
#pragma unroll
            for (int r = 0; r < 8; ++r) crow[(size_t)r * O] = c[r];
        } else {
#pragma unroll
            for (int r = 0; r < 8; ++r)
                if (cm0 + r < M) crow[(size_t)r * O] = c[r];
        }
    }
}

// ---------------------------------------------------------------- log_softmax over 40 classes
// acc is N x 48 (padded, bias already included); out is N x 40.
__global__ void logsoftmax40(const float* __restrict__ acc, float* __restrict__ out, int N) {
    int row  = blockIdx.x * (blockDim.x >> 5) + (threadIdx.x >> 5);
    int lane = threadIdx.x & 31;
    if (row >= N) return;
    const float* a = acc + (size_t)row * 48;
    float v0 = (lane < 40) ? a[lane] : -INFINITY;
    float v1 = (lane + 32 < 40) ? a[lane + 32] : -INFINITY;
    float m = fmaxf(v0, v1);
    for (int off = 16; off > 0; off >>= 1) m = fmaxf(m, __shfl_xor(m, off, 32));
    float s = ((lane < 40) ? __expf(v0 - m) : 0.0f) + ((lane + 32 < 40) ? __expf(v1 - m) : 0.0f);
    for (int off = 16; off > 0; off >>= 1) s += __shfl_xor(s, off, 32);
    float lse = m + __logf(s);
    float* o = out + (size_t)row * 40;
    if (lane < 40) o[lane] = v0 - lse;
    if (lane + 32 < 40) o[lane + 32] = v1 - lse;
}

// ---------------------------------------------------------------- launcher

extern "C" void kernel_launch(void* const* d_in, const int* in_sizes, int n_in,
                              void* d_out, int out_size, void* d_ws, size_t ws_size,
                              hipStream_t stream) {
    (void)n_in; (void)out_size; (void)ws_size;
    const float* x    = (const float*)d_in[0];
    const int*   esrc = (const int*)d_in[1];
    const int*   edst = (const int*)d_in[2];
    const float* W0   = (const float*)d_in[3];
    const float* b0   = (const float*)d_in[4];
    const float* W1   = (const float*)d_in[5];
    const float* b1   = (const float*)d_in[6];
    const float* W2   = (const float*)d_in[7];
    const float* b2   = (const float*)d_in[8];
    float* out = (float*)d_out;

    const int Nn = in_sizes[0] / 128;   // 50000
    const int E  = in_sizes[1];         // 800000
    const int F  = 128, K = 4, OP = 48; // OP: padded class dim (40 -> 48)
    const long NF = (long)Nn * F;

    // workspace carve (all 16B aligned: offsets are multiples of 4 floats)
    float* deg   = (float*)d_ws;
    float* dinv  = deg + Nn;
    float* wnorm = dinv + Nn;
    float* h     = wnorm + E;
    float* t1    = h  + NF;
    float* t2    = t1 + NF;
    float* t3    = t2 + NF;
    float* acc   = t3 + NF;
    _Float16* w0t = (_Float16*)(acc + NF);
    _Float16* w1t = w0t + (size_t)K * F * F;
    _Float16* w2t = w1t + (size_t)K * F * F;   // K * OP * F halfs

    const int TB = 256;
    auto blocks = [](long n, int tb) { return (unsigned)((n + tb - 1) / tb); };

    // --- edge normalization ---
    zero_f32<<<blocks(Nn, TB), TB, 0, stream>>>(deg, Nn);
    deg_count<<<blocks(E, TB), TB, 0, stream>>>(esrc, deg, E);
    dinv_kernel<<<blocks(Nn, TB), TB, 0, stream>>>(deg, dinv, Nn);
    wnorm_kernel<<<blocks(E, TB), TB, 0, stream>>>(esrc, edst, dinv, wnorm, E);

    // --- weights to f16 column-major ---
    convert_w<<<blocks((long)K * F * F, TB), TB, 0, stream>>>(W0, w0t, F, F, K * F * F);
    convert_w<<<blocks((long)K * F * F, TB), TB, 0, stream>>>(W1, w1t, F, F, K * F * F);
    convert_w<<<blocks((long)K * OP * F, TB), TB, 0, stream>>>(W2, w2t, 40, OP, K * OP * F);

    const unsigned gemmGrid = (unsigned)(((Nn + 15) / 16 + 7) / 8);
    auto spmm = [&](const float* X, float* Y) {
        spmm_edge<<<blocks((long)E * 32, TB), TB, 0, stream>>>(esrc, edst, wnorm, X, Y, E);
    };

    auto layer = [&](const float* hin, const _Float16* wt, int O,
                     const float* bias, int biasN) {
        auto g0 = [&](const float* Ap, const _Float16* Wp) {   // C = bias + A*W
            if (O == 128) cheb_gemm_wmma<128, true><<<gemmGrid, 256, 0, stream>>>(Ap, Wp, acc, bias, biasN, Nn);
            else          cheb_gemm_wmma< 48, true><<<gemmGrid, 256, 0, stream>>>(Ap, Wp, acc, bias, biasN, Nn);
        };
        auto g1 = [&](const float* Ap, const _Float16* Wp) {   // C += A*W
            if (O == 128) cheb_gemm_wmma<128, false><<<gemmGrid, 256, 0, stream>>>(Ap, Wp, acc, bias, biasN, Nn);
            else          cheb_gemm_wmma< 48, false><<<gemmGrid, 256, 0, stream>>>(Ap, Wp, acc, bias, biasN, Nn);
        };
        g0(hin, wt);                                             // k=0: Tx0 = hin (acc seeded with bias)
        zero_f32<<<blocks(NF, TB), TB, 0, stream>>>(t1, NF);
        spmm(hin, t1);                                           // Tx1 = L^ Tx0
        g1(t1, wt + (size_t)1 * O * F);
        zero_f32<<<blocks(NF, TB), TB, 0, stream>>>(t2, NF);
        spmm(t1, t2);
        fma2sub<<<blocks(NF, TB), TB, 0, stream>>>(t2, hin, NF); // Tx2 = 2 L^ Tx1 - Tx0
        g1(t2, wt + (size_t)2 * O * F);
        zero_f32<<<blocks(NF, TB), TB, 0, stream>>>(t3, NF);
        spmm(t2, t3);
        fma2sub<<<blocks(NF, TB), TB, 0, stream>>>(t3, t1, NF);  // Tx3 = 2 L^ Tx2 - Tx1
        g1(t3, wt + (size_t)3 * O * F);
    };

    // layer 0: x -> h
    layer(x, w0t, F, b0, F);
    relu_k<<<blocks(NF, TB), TB, 0, stream>>>(acc, h, NF);
    // layer 1: h -> h
    layer(h, w1t, F, b1, F);
    relu_k<<<blocks(NF, TB), TB, 0, stream>>>(acc, h, NF);
    // layer 2: h -> acc (N x 48, first 40 cols meaningful; bias folded, pad bias = 0)
    layer(h, w2t, OP, b2, 40);
    // log_softmax of pre-activation -> out (N x 40)
    logsoftmax40<<<blocks((long)Nn * 32, TB), TB, 0, stream>>>(acc, out, Nn);
}
